// APPNP_28991029248858
// MI455X (gfx1250) — matrix operand order repaired and checked by reading, not compiled, not run on hardware
//
#include <hip/hip_runtime.h>
#include <stdint.h>

#define N_NODES 100000
#define N_EDGES 1600000
#define IN_CH   512
#define HID_CH  256
#define OUT_CH  48
#define K_STEPS 10
#define ALPHA_F 0.1f

typedef _Float16 f16_t;
typedef __attribute__((ext_vector_type(8)))  _Float16 v8h;
typedef __attribute__((ext_vector_type(16))) _Float16 v16h;
typedef __attribute__((ext_vector_type(8)))  float    v8f;
typedef __attribute__((ext_vector_type(4)))  float    v4f;
typedef __attribute__((ext_vector_type(4)))  unsigned int v4u;
typedef __attribute__((ext_vector_type(8)))  int      v8i;
typedef __attribute__((ext_vector_type(4)))  int      v4i;

// ---------------------------------------------------------------------------
// CDNA5 async copy: global -> LDS, 16B per active lane, tracked by ASYNCcnt.
// ---------------------------------------------------------------------------
static __device__ __forceinline__ void async_copy_b128(f16_t* lds_dst,
                                                       const f16_t* gsrc) {
  const uint32_t ldsa = (uint32_t)(uintptr_t)lds_dst;
  asm volatile("global_load_async_to_lds_b128 %0, %1, off"
               :: "v"(ldsa), "v"(gsrc)
               : "memory");
}
static __device__ __forceinline__ void wait_async0() {
  asm volatile("s_wait_asynccnt 0x0" ::: "memory");
}

// ---------------------------------------------------------------------------
// CDNA5 Tensor Data Mover: one descriptor DMAs a 2D f16 tile (tile_x x tile_y,
// global row stride stride_x elems) into LDS, inserting pad_amt(+1) DWORDs of
// LDS padding after every 2^(pad_int+1) DWORDs (ISA 8.3/8.4 D# bitfields).
// This toolchain exposes the 6-arg builtin (g0, g1, g2, g3, g4, cpol).
// ---------------------------------------------------------------------------
static __device__ __forceinline__ void tdm_load_2d_f16(
    uint32_t lds_addr, const void* gaddr, uint32_t tile_x, uint32_t tile_y,
    uint32_t tensor_x, uint32_t tensor_y, uint32_t stride_x,
    uint32_t pad_en, uint32_t pad_int, uint32_t pad_amt) {
  const uint64_t ga = (uint64_t)(uintptr_t)gaddr;
  v4u g0;
  g0[0] = 1u;                                        // count=1 (valid), user mode
  g0[1] = lds_addr;                                  // LDS byte address
  g0[2] = (uint32_t)(ga & 0xFFFFFFFFu);              // global_addr[31:0]
  g0[3] = (uint32_t)((ga >> 32) & 0x01FFFFFFu)       // global_addr[56:32]
          | (2u << 30);                              // type = 2 ("image")
  v8i g1;
  g1[0] = (int)((1u << 16)                           // data_size = 1 (2 bytes)
                | (pad_en << 20) | (pad_int << 22) | (pad_amt << 25));
  g1[1] = (int)((tensor_x & 0xFFFFu) << 16);         // tensor_dim0[15:0]
  g1[2] = (int)(((tensor_x >> 16) & 0xFFFFu) | ((tensor_y & 0xFFFFu) << 16));
  g1[3] = (int)(((tensor_y >> 16) & 0xFFFFu) | ((tile_x & 0xFFFFu) << 16));
  g1[4] = (int)(tile_y & 0xFFFFu);                   // tile_dim1 (tile_dim2=0)
  g1[5] = (int)stride_x;                             // tensor_dim0_stride[31:0]
  g1[6] = 0;
  g1[7] = 0;
  v4i gz4 = {0, 0, 0, 0};                            // dims 2/3 unused (2D)
  v8i gz8 = {0, 0, 0, 0, 0, 0, 0, 0};
  __builtin_amdgcn_tensor_load_to_lds(g0, g1, gz4, gz4, gz8, 0);
}

// ---------------------------------------------------------------------------
// WMMA fragment loaders (CDNA5 16x16x32 f16, wave32 layouts per ISA 7.12.2)
// ---------------------------------------------------------------------------
static __device__ __forceinline__ v16h load_frag_a(const f16_t* base, int ld) {
  const int lane = threadIdx.x & 31;
  const f16_t* p = base + (lane & 15) * ld + ((lane >> 4) << 3);
  v8h lo = *(const v8h*)(p);
  v8h hi = *(const v8h*)(p + 16);
  v16h f;
#pragma unroll
  for (int i = 0; i < 8; ++i) { f[i] = lo[i]; f[i + 8] = hi[i]; }
  return f;
}

static __device__ __forceinline__ v16h load_frag_b(const f16_t* base, int ld) {
  const int lane = threadIdx.x & 31;
  const f16_t* p = base + (lane & 15) * ld + ((lane >> 4) << 3);
  v8h lo = *(const v8h*)(p);
  v8h hi = *(const v8h*)(p + 16);
  v16h f;
#pragma unroll
  for (int i = 0; i < 8; ++i) { f[i] = lo[i]; f[i + 8] = hi[i]; }
  return f;
}

#define WMMA_F16(A, B, C) \
  __builtin_amdgcn_wmma_f32_16x16x32_f16(false, (A), false, (B), (short)0, (C), false, false)

// ---------------------------------------------------------------------------
// GEMM1: H1[100000x256] (f16) = relu( X[100000x512] (f32) @ W1 + b1 )
// A: global f32 -> regs -> cvt -> LDS;  B: TDM tensor_load_to_lds (wave 0).
// Double-buffered LDS, 128 threads = 4 waves, 64x64 tile, K-tile 32.
// ---------------------------------------------------------------------------
#define G1_BM 64
#define G1_BN 64
#define G_BK  32
#define G_LD  40   // padded row stride (elems): 16B aligned, bank-conflict-free

__global__ __launch_bounds__(128) void gemm1_relu(
    const float* __restrict__ X, const f16_t* __restrict__ W1T,
    const float* __restrict__ B1, f16_t* __restrict__ H1) {
  __shared__ alignas(16) f16_t sA[2][G1_BM * G_LD];
  __shared__ alignas(16) f16_t sB[2][G1_BN * G_LD];
  const int tid = threadIdx.x, wave = tid >> 5, lane = tid & 31;
  const int m0 = blockIdx.x * G1_BM, n0 = blockIdx.y * G1_BN;

  v8f acc[4];
#pragma unroll
  for (int j = 0; j < 4; ++j)
#pragma unroll
    for (int r = 0; r < 8; ++r) acc[j][r] = 0.f;

  v4f pref[4];                       // A prefetch registers (16 f32/thread)

  auto issueA = [&](int k0) {        // global f32 loads -> registers
#pragma unroll
    for (int s = 0; s < 4; ++s) {
      const int i = tid + s * 128;   // 512 float4 chunks
      const int r = i >> 3, c = (i & 7) << 2;
      int gr = m0 + r;               // branchless clamp: OOB rows only feed
      gr = gr < N_NODES ? gr : N_NODES - 1;  // masked-off output rows
      pref[s] = *(const v4f*)(X + (size_t)gr * IN_CH + k0 + c);
    }
  };
  auto commitA = [&](int buf) {      // cvt f32->f16, store to LDS
#pragma unroll
    for (int s = 0; s < 4; ++s) {
      const int i = tid + s * 128;
      const int r = i >> 3, c = (i & 7) << 2;
      f16_t* d = &sA[buf][r * G_LD + c];
      d[0] = (f16_t)pref[s][0]; d[1] = (f16_t)pref[s][1];
      d[2] = (f16_t)pref[s][2]; d[3] = (f16_t)pref[s][3];
    }
  };
  auto stageB = [&](int buf, int k0) {  // TDM: 64x32 f16 tile, LDS-padded to 40
    if (wave == 0) {
      tdm_load_2d_f16((uint32_t)(uintptr_t)&sB[buf][0],
                      W1T + (size_t)n0 * IN_CH + k0,
                      /*tile_x=*/G_BK, /*tile_y=*/G1_BN,
                      /*tensor_x=*/IN_CH, /*tensor_y=*/HID_CH,
                      /*stride_x=*/IN_CH,
                      /*pad_en=*/1, /*pad_int=*/3, /*pad_amt=*/3);
    }
  };

  const int T = IN_CH / G_BK;        // 16 K-tiles
  issueA(0);
  stageB(0, 0);
  commitA(0);

  for (int t = 0; t < T; ++t) {
    const int buf = t & 1;
    __builtin_amdgcn_s_wait_tensorcnt(0);   // no-op for waves 1..3
    __syncthreads();
    if (t + 1 < T) {                 // overlap next tile's traffic with WMMA
      issueA((t + 1) * G_BK);
      stageB(buf ^ 1, (t + 1) * G_BK);
    }
    v16h fa  = load_frag_a(&sA[buf][(wave << 4) * G_LD], G_LD);
    v16h fb0 = load_frag_b(&sB[buf][0 * 16 * G_LD], G_LD);
    v16h fb1 = load_frag_b(&sB[buf][1 * 16 * G_LD], G_LD);
    v16h fb2 = load_frag_b(&sB[buf][2 * 16 * G_LD], G_LD);
    v16h fb3 = load_frag_b(&sB[buf][3 * 16 * G_LD], G_LD);
    acc[0] = WMMA_F16(fa, fb0, acc[0]);
    acc[1] = WMMA_F16(fa, fb1, acc[1]);
    acc[2] = WMMA_F16(fa, fb2, acc[2]);
    acc[3] = WMMA_F16(fa, fb3, acc[3]);
    if (t + 1 < T) commitA(buf ^ 1);
  }

  const int colL = lane & 15, rowH = (lane >> 4) << 3;
#pragma unroll
  for (int j = 0; j < 4; ++j) {
    const int n = n0 + (j << 4) + colL;
    const float bias = B1[n];
#pragma unroll
    for (int r = 0; r < 8; ++r) {
      const int m = m0 + (wave << 4) + rowH + r;
      if (m < N_NODES) {
        float v = acc[j][r] + bias;
        H1[(size_t)m * HID_CH + n] = (f16_t)(v > 0.f ? v : 0.f);
      }
    }
  }
}

// ---------------------------------------------------------------------------
// GEMM2: H[100000x48] (f32) = H1[100000x256] (f16) @ W2 + b2
// Both A and B tiles staged via per-lane async-to-LDS (all-f16 sources).
// ---------------------------------------------------------------------------
#define G2_BM 64
#define G2_BN 48

__global__ __launch_bounds__(128) void gemm2(
    const f16_t* __restrict__ H1, const f16_t* __restrict__ W2T,
    const float* __restrict__ B2, float* __restrict__ H) {
  __shared__ alignas(16) f16_t sA[2][G2_BM * G_LD];
  __shared__ alignas(16) f16_t sB[2][G2_BN * G_LD];
  const int tid = threadIdx.x, wave = tid >> 5, lane = tid & 31;
  const int m0 = blockIdx.x * G2_BM;

  v8f acc[3];
#pragma unroll
  for (int j = 0; j < 3; ++j)
#pragma unroll
    for (int r = 0; r < 8; ++r) acc[j][r] = 0.f;

  auto stageA = [&](int buf, int k0) {
#pragma unroll
    for (int s = 0; s < 2; ++s) {
      const int i = tid + s * 128;   // 256 chunks
      const int r = i >> 2, c = (i & 3) << 3;
      int gr = m0 + r;               // branchless clamp (see GEMM1)
      gr = gr < N_NODES ? gr : N_NODES - 1;
      async_copy_b128(&sA[buf][r * G_LD + c],
                      H1 + (size_t)gr * HID_CH + k0 + c);
    }
  };
  auto stageB = [&](int buf, int k0) {
#pragma unroll
    for (int s = 0; s < 2; ++s) {
      const int i = tid + s * 128;   // 192 chunks
      if (i < G2_BN * 4) {
        const int r = i >> 2, c = (i & 3) << 3;
        async_copy_b128(&sB[buf][r * G_LD + c],
                        W2T + (size_t)r * HID_CH + k0 + c);
      }
    }
  };

  const int T = HID_CH / G_BK;       // 8 K-tiles
  stageA(0, 0);
  stageB(0, 0);

  for (int t = 0; t < T; ++t) {
    const int buf = t & 1;
    wait_async0();
    __syncthreads();
    if (t + 1 < T) {
      stageA(buf ^ 1, (t + 1) * G_BK);
      stageB(buf ^ 1, (t + 1) * G_BK);
    }
    v16h fa  = load_frag_a(&sA[buf][(wave << 4) * G_LD], G_LD);
    v16h fb0 = load_frag_b(&sB[buf][0 * 16 * G_LD], G_LD);
    v16h fb1 = load_frag_b(&sB[buf][1 * 16 * G_LD], G_LD);
    v16h fb2 = load_frag_b(&sB[buf][2 * 16 * G_LD], G_LD);
    acc[0] = WMMA_F16(fa, fb0, acc[0]);
    acc[1] = WMMA_F16(fa, fb1, acc[1]);
    acc[2] = WMMA_F16(fa, fb2, acc[2]);
  }

  const int colL = lane & 15, rowH = (lane >> 4) << 3;
#pragma unroll
  for (int j = 0; j < 3; ++j) {
    const int n = (j << 4) + colL;
    const float bias = B2[n];
#pragma unroll
    for (int r = 0; r < 8; ++r) {
      const int m = m0 + (wave << 4) + rowH + r;
      if (m < N_NODES) H[(size_t)m * OUT_CH + n] = acc[j][r] + bias;
    }
  }
}

// ---------------------------------------------------------------------------
// Weight transpose + f32->f16 convert (tiny, one-shot)
// ---------------------------------------------------------------------------
__global__ void prep_w1t(const float* __restrict__ W1, f16_t* __restrict__ W1T) {
  const int i = blockIdx.x * 256 + threadIdx.x;   // over IN_CH*HID_CH
  if (i < IN_CH * HID_CH) {
    const int k = i / HID_CH, n = i % HID_CH;     // W1[k][n]
    W1T[(size_t)n * IN_CH + k] = (f16_t)W1[i];
  }
}
__global__ void prep_w2t(const float* __restrict__ W2, f16_t* __restrict__ W2T) {
  const int i = blockIdx.x * 256 + threadIdx.x;   // over HID_CH*OUT_CH
  if (i < HID_CH * OUT_CH) {
    const int k = i / OUT_CH, n = i % OUT_CH;     // W2[k][n]
    W2T[(size_t)n * HID_CH + k] = (f16_t)W2[i];
  }
}

// ---------------------------------------------------------------------------
// Degree / normalization: deg[i] = 1 (self loop) + #(dst==i); dinv = rsqrt
// ---------------------------------------------------------------------------
__global__ void deg_init(float* __restrict__ deg) {
  const int i = blockIdx.x * 256 + threadIdx.x;
  if (i < N_NODES) deg[i] = 1.0f;
}
__global__ void deg_edges(const int* __restrict__ dst, float* __restrict__ deg) {
  const int e = blockIdx.x * 256 + threadIdx.x;
  if (e < N_EDGES) atomicAdd(&deg[dst[e]], 1.0f);
}
__global__ void deg_finish(float* __restrict__ deg) {
  const int i = blockIdx.x * 256 + threadIdx.x;
  if (i < N_NODES) deg[i] = rsqrtf(deg[i]);   // deg >= 1 always
}

// ---------------------------------------------------------------------------
// Propagation step: out = (1-a)*(self-loop + edge scatter of w*z[src]) + a*h
// Working set (z, out, h, edges, dinv) is L2-resident on MI455X (192 MB L2).
// ---------------------------------------------------------------------------
__global__ void prop_init(const float* __restrict__ h, const float* __restrict__ z,
                          const float* __restrict__ dinv, float* __restrict__ out) {
  const int i = blockIdx.x * 256 + threadIdx.x;   // over N_NODES*OUT_CH
  if (i < N_NODES * OUT_CH) {
    const int node = i / OUT_CH;
    const float di = dinv[node];
    out[i] = ALPHA_F * h[i] + (1.0f - ALPHA_F) * di * di * z[i];
  }
}

__global__ void prop_edges(const int* __restrict__ src, const int* __restrict__ dst,
                           const float* __restrict__ dinv, const float* __restrict__ z,
                           float* __restrict__ out) {
  const int t = blockIdx.x * 256 + threadIdx.x;
  const int e = t >> 4;          // 16 lanes cooperate on one edge (48 channels)
  const int l = t & 15;
  if (e < N_EDGES) {
    const int s = src[e], d = dst[e];
    const float c = (1.0f - ALPHA_F) * dinv[s] * dinv[d];
    const float* zs = z + (size_t)s * OUT_CH;
    float* od = out + (size_t)d * OUT_CH;
    atomicAdd(&od[l],      c * zs[l]);
    atomicAdd(&od[l + 16], c * zs[l + 16]);
    atomicAdd(&od[l + 32], c * zs[l + 32]);
  }
}

// ---------------------------------------------------------------------------
// Host launcher
// ---------------------------------------------------------------------------
extern "C" void kernel_launch(void* const* d_in, const int* in_sizes, int n_in,
                              void* d_out, int out_size, void* d_ws, size_t ws_size,
                              hipStream_t stream) {
  (void)in_sizes; (void)n_in; (void)out_size; (void)ws_size;
  const float* X  = (const float*)d_in[0];
  const int*   EI = (const int*)d_in[1];
  const float* W1 = (const float*)d_in[2];
  const float* B1 = (const float*)d_in[3];
  const float* W2 = (const float*)d_in[4];
  const float* B2 = (const float*)d_in[5];
  float* OUT = (float*)d_out;

  char* ws = (char*)d_ws;
  size_t off = 0;
  auto alloc = [&](size_t bytes) -> void* {
    void* p = (void*)(ws + off);
    off += (bytes + 255) & ~(size_t)255;
    return p;
  };
  float* dinv = (float*)alloc(sizeof(float) * N_NODES);
  f16_t* W1T  = (f16_t*)alloc(sizeof(f16_t) * (size_t)IN_CH * HID_CH);
  f16_t* W2T  = (f16_t*)alloc(sizeof(f16_t) * (size_t)HID_CH * OUT_CH);
  f16_t* H1   = (f16_t*)alloc(sizeof(f16_t) * (size_t)N_NODES * HID_CH);
  float* H    = (float*)alloc(sizeof(float) * (size_t)N_NODES * OUT_CH);
  float* ZA   = (float*)alloc(sizeof(float) * (size_t)N_NODES * OUT_CH);
  float* ZB   = (float*)alloc(sizeof(float) * (size_t)N_NODES * OUT_CH);

  const int* SRC = EI;
  const int* DST = EI + N_EDGES;

  prep_w1t<<<(IN_CH * HID_CH + 255) / 256, 256, 0, stream>>>(W1, W1T);
  prep_w2t<<<(HID_CH * OUT_CH + 255) / 256, 256, 0, stream>>>(W2, W2T);

  deg_init  <<<(N_NODES + 255) / 256, 256, 0, stream>>>(dinv);
  deg_edges <<<(N_EDGES + 255) / 256, 256, 0, stream>>>(DST, dinv);
  deg_finish<<<(N_NODES + 255) / 256, 256, 0, stream>>>(dinv);

  gemm1_relu<<<dim3((N_NODES + G1_BM - 1) / G1_BM, HID_CH / G1_BN), 128, 0, stream>>>(
      X, W1T, B1, H1);
  gemm2<<<dim3((N_NODES + G2_BM - 1) / G2_BM, 1), 128, 0, stream>>>(
      H1, W2T, B2, H);

  const float* zin = H;
  for (int k = 0; k < K_STEPS; ++k) {
    float* zout = (k == K_STEPS - 1) ? OUT : ((k & 1) ? ZB : ZA);
    prop_init <<<(N_NODES * OUT_CH + 255) / 256, 256, 0, stream>>>(H, zin, dinv, zout);
    prop_edges<<<(N_EDGES * 16 + 255) / 256, 256, 0, stream>>>(SRC, DST, dinv, zin, zout);
    zin = zout;
  }
}
